// LocalPatchEncoder_11407433138451
// MI455X (gfx1250) — compile-verified, matrix operand order checked by reading, not run on hardware
//
#include <hip/hip_runtime.h>
#include <hip/hip_bf16.h>
#include <hip/hip_fp16.h>

typedef __attribute__((ext_vector_type(16))) _Float16 v16h;
typedef __attribute__((ext_vector_type(8)))  float    v8f;
typedef __attribute__((ext_vector_type(4)))  float    v4f;

#define BATCH   8
#define NPTS    16384
#define NSEG    1024
#define KNBR    32
#define C_REAL  91
#define C_IN    96        // 91 padded to 96 (3 x K=32 WMMA steps)
#define C_OUT   256
#define LNEPS   1e-5f
#define RAD2    0.0625f   // 0.25^2

// ---------------------------------------------------------------------------
// Kernel 1: ball query. One wave per query; chunked in-order scan with ballot
// + prefix popcount -> first 32 smallest-index points within radius, early exit.
// ---------------------------------------------------------------------------
__global__ void k_neighbors(const float* __restrict__ xyz,
                            const float* __restrict__ ctr,
                            int* __restrict__ out_idx) {
  __shared__ int nbuf[8][KNBR];
  const int w    = threadIdx.x >> 5;
  const int lane = threadIdx.x & 31;
  const int q    = blockIdx.x * 8 + w;          // 0..8191
  const int b    = q >> 10;
  const int s    = q & 1023;

  const float cx = ctr[(b * NSEG + s) * 3 + 0];
  const float cy = ctr[(b * NSEG + s) * 3 + 1];
  const float cz = ctr[(b * NSEG + s) * 3 + 2];
  const float cs = cx * cx + cy * cy + cz * cz;
  const float* pb = xyz + (size_t)b * NPTS * 3;

  int cnt = 0;
  for (int base = 0; base < NPTS; base += 32) {
    const int i = base + lane;
    const float px = pb[i * 3 + 0];
    const float py = pb[i * 3 + 1];
    const float pz = pb[i * 3 + 2];
    // match reference formula: |c|^2 + |p|^2 - 2 c.p  (exclude if > r^2)
    const float sqr = cs + px * px + py * py + pz * pz
                    - 2.0f * (cx * px + cy * py + cz * pz);
    const bool hit = !(sqr > RAD2);
    const unsigned m = __builtin_amdgcn_ballot_w32(hit);
    if (hit) {
      const int pos = cnt + __popc(m & ((1u << lane) - 1u));
      if (pos < KNBR) nbuf[w][pos] = i;
    }
    cnt += __popc(m);
    if (cnt >= KNBR) break;                      // wave-uniform
  }
  __syncthreads();                               // every thread reaches once
  int v;
  if (lane < cnt) v = nbuf[w][lane];
  else            v = (cnt > 0) ? nbuf[w][0] : 0;   // pad with first index
  out_idx[q * KNBR + lane] = v;
}

// ---------------------------------------------------------------------------
// Kernel 2: convert weights to f16 and transpose to [N][K] (column-major for
// the GEMM) so B-fragments become two contiguous 16B loads per lane.
// W1t: [256][96] with K padded 91->96 (zeros). W2t: [256][256].
// ---------------------------------------------------------------------------
__global__ void k_weights(const float* __restrict__ W1,
                          const float* __restrict__ W2,
                          _Float16* __restrict__ w1t,
                          _Float16* __restrict__ w2t) {
  const int i = blockIdx.x * blockDim.x + threadIdx.x;
  const int tot1 = C_OUT * C_IN;
  if (i < tot1) {
    const int n = i / C_IN, k = i % C_IN;
    w1t[i] = (k < C_REAL) ? (_Float16)W1[k * C_OUT + n] : (_Float16)0.f;
  } else if (i < tot1 + C_OUT * C_OUT) {
    const int j = i - tot1;
    const int n = j >> 8, k = j & 255;
    w2t[j] = (_Float16)W2[k * C_OUT + n];
  }
}

// Fragment loader matching the CDNA5 16-bit A/B VGPR layout (wave32):
// element e of lane L sits at K = (e&7) + 8*(L>=16) + 16*(e>=8)  ->
// two contiguous 8-half runs -> two 128-bit loads.
__device__ __forceinline__ v16h load_frag(const _Float16* rowbase, int kk, int lane) {
  const int b8 = (lane >> 4) << 3;
  union { v16h v; v4f f[2]; } u;
  u.f[0] = *(const v4f*)(rowbase + kk + b8);
  u.f[1] = *(const v4f*)(rowbase + kk + b8 + 16);
  return u.v;
}

// ---------------------------------------------------------------------------
// Kernel 3: fused per-segment pipeline. Block = 512 threads = 16 waves.
// gather+encode -> WMMA GEMM1 -> LN+GELU -> WMMA GEMM2 -> LN -> max over K.
// ---------------------------------------------------------------------------
__global__ void __launch_bounds__(512) k_main(
    const float* __restrict__ xyz, const float* __restrict__ pf,
    const float* __restrict__ ctr,
    const _Float16* __restrict__ w1t,
    const float* __restrict__ b1, const float* __restrict__ g1,
    const float* __restrict__ be1,
    const _Float16* __restrict__ w2t,
    const float* __restrict__ b2, const float* __restrict__ g2,
    const float* __restrict__ be2,
    const int* __restrict__ nb_all,
    float* __restrict__ out_feat) {
  __shared__ int nidx[KNBR];
  __shared__ __align__(16) _Float16 sA [KNBR][C_IN];    //  6 KB input tile
  __shared__ __align__(16) _Float16 h1h[KNBR][C_OUT];   // 16 KB layer1 (f16)
  __shared__ __align__(16) float    hf [KNBR][C_OUT];   // 32 KB f32 staging

  const int tid = threadIdx.x;
  const int seg = blockIdx.x;
  const int b   = seg >> 10;
  const int s   = seg & 1023;

  if (tid < KNBR) nidx[tid] = nb_all[seg * KNBR + tid];
  __syncthreads();

  // --- build input rows: [feat(64) | rel_xyz(3) | posenc(24) | pad(5)] -----
  for (int i = tid; i < KNBR * 64; i += 512) {
    const int row = i >> 6, c = i & 63;
    sA[row][c] = (_Float16)pf[((size_t)b * NPTS + nidx[row]) * 64 + c];
  }
  if (tid < KNBR * 3) {
    const int row = tid / 3, d = tid % 3;
    const float rel = xyz[((size_t)b * NPTS + nidx[row]) * 3 + d]
                    - ctr[((size_t)b * NSEG + s) * 3 + d];
    sA[row][64 + d] = (_Float16)rel;
    float f = 1.f;
    for (int j = 0; j < 4; ++j) {                  // freqs 1,2,4,8
      sA[row][67 + d * 8 + j]     = (_Float16)__sinf(rel * f);
      sA[row][67 + d * 8 + 4 + j] = (_Float16)__cosf(rel * f);
      f *= 2.f;
    }
  }
  if (tid < KNBR * 5) {                            // zero the K-pad 91..95
    sA[tid / 5][C_REAL + tid % 5] = (_Float16)0.f;
  }
  __syncthreads();

  const int w    = tid >> 5;
  const int lane = tid & 31;
  const int nn   = w * 16 + (lane & 15);           // output column
  const int m0   = lane & 15;                      // A-frag row (M-tile 0)
  const v8f zero = {};

  // --- GEMM1: [32 x 96] x [96 x 256] (two 16-row M-tiles per wave) ---------
  v8f acc0 = zero, acc1 = zero;
  {
    const _Float16* bcol = w1t + nn * C_IN;
    for (int kk = 0; kk < C_IN; kk += 32) {
      const v16h bf = load_frag(bcol, kk, lane);
      const v16h a0 = load_frag(&sA[m0][0],      kk, lane);
      const v16h a1 = load_frag(&sA[m0 + 16][0], kk, lane);
      acc0 = __builtin_amdgcn_wmma_f32_16x16x32_f16(false, a0, false, bf,
                                                    (short)0, acc0, false, false);
      acc1 = __builtin_amdgcn_wmma_f32_16x16x32_f16(false, a1, false, bf,
                                                    (short)0, acc1, false, false);
    }
  }
  {
    const float bias = b1[nn];
    for (int r = 0; r < 8; ++r) {
      const int mr = r + ((lane >> 4) << 3);       // C/D layout: lanes>=16 -> M+8
      hf[mr][nn]      = acc0[r] + bias;
      hf[mr + 16][nn] = acc1[r] + bias;
    }
  }
  __syncthreads();

  // --- LayerNorm + exact GELU -> h1h (f16). Wave w owns rows w, w+16 -------
  for (int rr = 0; rr < 2; ++rr) {
    const int row = w + rr * 16;
    float x[8], sum = 0.f, sq = 0.f;
    for (int t = 0; t < 8; ++t) {
      x[t] = hf[row][lane + t * 32];
      sum += x[t]; sq += x[t] * x[t];
    }
    for (int off = 16; off > 0; off >>= 1) {
      sum += __shfl_xor(sum, off, 32);
      sq  += __shfl_xor(sq,  off, 32);
    }
    const float mu = sum * (1.f / 256.f);
    const float rs = rsqrtf(sq * (1.f / 256.f) - mu * mu + LNEPS);
    for (int t = 0; t < 8; ++t) {
      const int c = lane + t * 32;
      const float y = (x[t] - mu) * rs * g1[c] + be1[c];
      h1h[row][c] = (_Float16)(0.5f * y * (1.f + erff(y * 0.70710678118654752f)));
    }
  }
  __syncthreads();

  // --- GEMM2: [32 x 256] x [256 x 256] -------------------------------------
  acc0 = zero; acc1 = zero;
  {
    const _Float16* bcol = w2t + nn * C_OUT;
    for (int kk = 0; kk < C_OUT; kk += 32) {
      const v16h bf = load_frag(bcol, kk, lane);
      const v16h a0 = load_frag(&h1h[m0][0],      kk, lane);
      const v16h a1 = load_frag(&h1h[m0 + 16][0], kk, lane);
      acc0 = __builtin_amdgcn_wmma_f32_16x16x32_f16(false, a0, false, bf,
                                                    (short)0, acc0, false, false);
      acc1 = __builtin_amdgcn_wmma_f32_16x16x32_f16(false, a1, false, bf,
                                                    (short)0, acc1, false, false);
    }
  }
  {
    const float bias = b2[nn];
    for (int r = 0; r < 8; ++r) {
      const int mr = r + ((lane >> 4) << 3);
      hf[mr][nn]      = acc0[r] + bias;
      hf[mr + 16][nn] = acc1[r] + bias;
    }
  }
  __syncthreads();

  // --- LayerNorm (no activation), in place ---------------------------------
  for (int rr = 0; rr < 2; ++rr) {
    const int row = w + rr * 16;
    float x[8], sum = 0.f, sq = 0.f;
    for (int t = 0; t < 8; ++t) {
      x[t] = hf[row][lane + t * 32];
      sum += x[t]; sq += x[t] * x[t];
    }
    for (int off = 16; off > 0; off >>= 1) {
      sum += __shfl_xor(sum, off, 32);
      sq  += __shfl_xor(sq,  off, 32);
    }
    const float mu = sum * (1.f / 256.f);
    const float rs = rsqrtf(sq * (1.f / 256.f) - mu * mu + LNEPS);
    for (int t = 0; t < 8; ++t) {
      const int c = lane + t * 32;
      hf[row][c] = (x[t] - mu) * rs * g2[c] + be2[c];
    }
  }
  __syncthreads();

  // --- max over the 32 neighbors -------------------------------------------
  if (tid < C_OUT) {
    float m = hf[0][tid];
    for (int r = 1; r < KNBR; ++r) m = fmaxf(m, hf[r][tid]);
    out_feat[(size_t)seg * C_OUT + tid] = m;
  }
}

// ---------------------------------------------------------------------------
extern "C" void kernel_launch(void* const* d_in, const int* in_sizes, int n_in,
                              void* d_out, int out_size, void* d_ws, size_t ws_size,
                              hipStream_t stream) {
  (void)in_sizes; (void)n_in; (void)out_size; (void)ws_size;
  const float* xyz = (const float*)d_in[0];
  const float* pf  = (const float*)d_in[1];
  const float* ctr = (const float*)d_in[2];
  const float* W1  = (const float*)d_in[3];
  const float* b1  = (const float*)d_in[4];
  const float* g1  = (const float*)d_in[5];
  const float* be1 = (const float*)d_in[6];
  const float* W2  = (const float*)d_in[7];
  const float* b2  = (const float*)d_in[8];
  const float* g2  = (const float*)d_in[9];
  const float* be2 = (const float*)d_in[10];

  float* out_feat = (float*)d_out;                              // (8,1024,256) f32
  int*   out_idx  = (int*)d_out + (size_t)BATCH * NSEG * C_OUT; // (8,1024,32) i32 bits

  _Float16* w1t = (_Float16*)d_ws;                  // [256][96]
  _Float16* w2t = w1t + C_OUT * C_IN;               // [256][256]

  k_neighbors<<<BATCH * NSEG / 8, 256, 0, stream>>>(xyz, ctr, out_idx);

  const int wtot = C_OUT * C_IN + C_OUT * C_OUT;
  k_weights<<<(wtot + 255) / 256, 256, 0, stream>>>(W1, W2, w1t, w2t);

  k_main<<<BATCH * NSEG, 512, 0, stream>>>(xyz, pf, ctr,
                                           w1t, b1, g1, be1,
                                           w2t, b2, g2, be2,
                                           out_idx, out_feat);
}
